// EncoderLayer_43344809951522
// MI455X (gfx1250) — matrix operand order repaired and checked by reading, not compiled, not run on hardware
//
#include <hip/hip_runtime.h>
#include <stdint.h>

typedef unsigned short u16;
typedef __attribute__((ext_vector_type(16))) __bf16 v16bf;
typedef __attribute__((ext_vector_type(8)))  float  v8f;
typedef __attribute__((ext_vector_type(4)))  int    v4i;

// address-space-qualified element types for the async-LDS builtin
typedef __attribute__((address_space(1))) v4i as1_v4i;
typedef __attribute__((address_space(3))) v4i as3_v4i;

union FragAB { v16bf v; uint4 u[2]; };

#define D_MODEL 1024
#define SEQ     2048
#define BATCH   2
#define NHEAD   16
#define DEPTH   64
#define DFF     4096
#define MROWS   (BATCH*SEQ)   // 4096

#if __has_builtin(__builtin_amdgcn_global_load_async_to_lds_b128) && \
    __has_builtin(__builtin_amdgcn_s_wait_asynccnt)
#define USE_ASYNC 1
#else
#define USE_ASYNC 0
#endif

__device__ __forceinline__ u16 f2bf(float f) {
    union { float f; unsigned u; } a; a.f = f;
    unsigned u = a.u;
    unsigned r = u + 0x7FFFu + ((u >> 16) & 1u);      // RNE
    if ((u & 0x7F800000u) == 0x7F800000u) r = u;      // Inf/NaN passthrough
    return (u16)(r >> 16);
}

__device__ __forceinline__ v8f wmma_bf16(const FragAB& a, const FragAB& b, v8f c) {
    return __builtin_amdgcn_wmma_f32_16x16x32_bf16(
        /*neg_a=*/false, a.v, /*neg_b=*/false, b.v,
        /*c_mod=*/(short)0, c, /*reuse_a=*/false, /*reuse_b=*/false);
}

// 16B global -> LDS copy; async (ASYNCcnt-tracked, LDS-direct) when available.
__device__ __forceinline__ void cp16_g2l(const u16* g, u16* l) {
#if USE_ASYNC
    __builtin_amdgcn_global_load_async_to_lds_b128(
        (as1_v4i*)(g), (as3_v4i*)(l), 0, 0);
#else
    *(uint4*)l = *(const uint4*)g;
#endif
}
__device__ __forceinline__ void cp_wait() {
#if USE_ASYNC
    __builtin_amdgcn_s_wait_asynccnt(0);
#endif
}

// ---------------------------------------------------------------- prep kernels

__global__ void cvt_bf16_kernel(const float* __restrict__ src, u16* __restrict__ dst, int n) {
    int i = blockIdx.x * blockDim.x + threadIdx.x;
    if (i < n) dst[i] = f2bf(src[i]);
}

// W [K][N] fp32 row-major  ->  WT [N][K] bf16 row-major
__global__ void transpose_w_kernel(const float* __restrict__ src, u16* __restrict__ dst, int K, int N) {
    int i = blockIdx.x * blockDim.x + threadIdx.x;
    if (i >= K * N) return;
    int k = i % K, n = i / K;
    dst[(size_t)n * K + k] = f2bf(src[(size_t)k * N + n]);
}

// v bf16 [B*S, H*64]  ->  vT bf16 [B, H, 64, S]
__global__ void transpose_v_kernel(const u16* __restrict__ vb, u16* __restrict__ vt) {
    int i = blockIdx.x * blockDim.x + threadIdx.x;
    if (i >= BATCH * NHEAD * DEPTH * SEQ) return;
    int s = i & (SEQ - 1);
    int d = (i >> 11) & (DEPTH - 1);
    int h = (i >> 17) & (NHEAD - 1);
    int b = i >> 21;
    vt[i] = vb[((size_t)(b * SEQ + s)) * D_MODEL + h * DEPTH + d];
}

// ---------------------------------------------------------------- GEMM
// C[M,N] = act( (A[M,K](bf16) @ B[K,N]) + bias ) * scale
// B supplied transposed: BT[N][K] bf16.
// Block tile 128x256, 8 waves of 64x64, K-stage 32, double-buffered LDS with
// async global->LDS copies overlapping the WMMA stream.

#define BM 128
#define BN 256
#define BK 32
#define LDSK 40   // padded K-stride (elements), 80B = 5*16B

__global__ __launch_bounds__(256) void gemm_bf16_kernel(
    const u16* __restrict__ A, const u16* __restrict__ BT, const float* __restrict__ bias,
    void* __restrict__ Cout, int M, int N, int K, float scale, int relu, int out_bf16)
{
    __shared__ __align__(16) u16 sA[2][BM * LDSK];
    __shared__ __align__(16) u16 sB[2][BN * LDSK];

    const int tid  = threadIdx.x;
    const int wave = tid >> 5, lane = tid & 31;
    const int wm = wave >> 2, wn = wave & 3;       // 2x4 wave grid
    const int bm = blockIdx.y * BM, bn = blockIdx.x * BN;
    const int lm = lane & 15, half = lane >> 4;

    v8f acc[4][4];
    const v8f vzero = {0.f,0.f,0.f,0.f,0.f,0.f,0.f,0.f};
#pragma unroll
    for (int i = 0; i < 4; ++i)
#pragma unroll
        for (int j = 0; j < 4; ++j) acc[i][j] = vzero;

    // stage one 128x32 A tile + 256x32 B tile into LDS buffer `buf`
    auto stage = [&](int buf, int k0) {
#pragma unroll
        for (int i = 0; i < 2; ++i) {              // A: 512 16B chunks
            int c = tid + i * 256, row = c >> 2, off = (c & 3) * 8;
            cp16_g2l(A + (size_t)(bm + row) * K + k0 + off, sA[buf] + row * LDSK + off);
        }
#pragma unroll
        for (int i = 0; i < 4; ++i) {              // BT: 1024 16B chunks
            int c = tid + i * 256, row = c >> 2, off = (c & 3) * 8;
            cp16_g2l(BT + (size_t)(bn + row) * K + k0 + off, sB[buf] + row * LDSK + off);
        }
    };

    const int nstages = K / BK;
    stage(0, 0);
    cp_wait();
    __syncthreads();

    for (int s = 0; s < nstages; ++s) {
        const int cur = s & 1;
        if (s + 1 < nstages) stage(cur ^ 1, (s + 1) * BK);   // prefetch next tile

        FragAB af[4], bf[4];
#pragma unroll
        for (int i = 0; i < 4; ++i) {
            const u16* base = sA[cur] + (wm * 64 + 16 * i + lm) * LDSK;
            af[i].u[0] = *(const uint4*)(base + half * 8);
            af[i].u[1] = *(const uint4*)(base + 16 + half * 8);
        }
#pragma unroll
        for (int j = 0; j < 4; ++j) {
            const u16* base = sB[cur] + (wn * 64 + 16 * j + lm) * LDSK;
            bf[j].u[0] = *(const uint4*)(base + half * 8);
            bf[j].u[1] = *(const uint4*)(base + 16 + half * 8);
        }
#pragma unroll
        for (int i = 0; i < 4; ++i)
#pragma unroll
            for (int j = 0; j < 4; ++j)
                acc[i][j] = wmma_bf16(af[i], bf[j], acc[i][j]);

        cp_wait();         // next tile fully landed in LDS
        __syncthreads();   // all waves done reading `cur` before it is restaged
    }

    // epilogue: C layout -> lane holds col lm of tile j, rows r + 8*half
#pragma unroll
    for (int i = 0; i < 4; ++i) {
#pragma unroll
        for (int j = 0; j < 4; ++j) {
            int col = bn + wn * 64 + 16 * j + lm;
            float bv = bias ? bias[col] : 0.f;
#pragma unroll
            for (int r = 0; r < 8; ++r) {
                int row = bm + wm * 64 + 16 * i + r + 8 * half;
                float v = (acc[i][j][r] + bv) * scale;
                if (relu) v = v > 0.f ? v : 0.f;
                if (out_bf16) ((u16*)Cout)[(size_t)row * N + col] = f2bf(v);
                else          ((float*)Cout)[(size_t)row * N + col] = v;
            }
        }
    }
}

// ---------------------------------------------------------------- attention
// q,k bf16 [B*S, H*64] (q pre-scaled by 1/8), vT bf16 [B,H,64,S].
// Flash-style: each wave owns 16 q rows; loop over 128-key blocks.
// ctx written bf16 [B*S, H*64] (== [B,S,D_MODEL]).

__global__ __launch_bounds__(256) void attn_kernel(
    const u16* __restrict__ qb, const u16* __restrict__ kb,
    const u16* __restrict__ vt, u16* __restrict__ ctx)
{
    __shared__ __align__(16) u16 P[8][16 * 136];   // per-wave 16x128 probs (bf16), padded

    const int tid = threadIdx.x;
    const int w = tid >> 5, lane = tid & 31;
    const int lm = lane & 15, half = lane >> 4;
    const int bh = blockIdx.x;
    const int b = bh >> 4, h = bh & 15;
    const int q0 = blockIdx.y * 128 + w * 16;

    // Q fragments (16 rows x 64 depth = 2 A-frags), kept in registers
    FragAB qf[2];
#pragma unroll
    for (int kc = 0; kc < 2; ++kc) {
        const u16* base = qb + (size_t)(b * SEQ + q0 + lm) * D_MODEL + h * DEPTH + kc * 32;
        qf[kc].u[0] = *(const uint4*)(base + half * 8);
        qf[kc].u[1] = *(const uint4*)(base + 16 + half * 8);
    }

    float mrow[8], lrow[8], alpha[8];
    v8f accc[4];
    const v8f vzero = {0.f,0.f,0.f,0.f,0.f,0.f,0.f,0.f};
#pragma unroll
    for (int r = 0; r < 8; ++r) { mrow[r] = -3.0e38f; lrow[r] = 0.f; }
#pragma unroll
    for (int j = 0; j < 4; ++j) accc[j] = vzero;

    u16* pl = P[w];

    for (int kblk = 0; kblk < SEQ; kblk += 128) {
        // ---- S = Q @ K^T  (16 x 128), 8 col tiles x 2 k-steps
        v8f s[8];
#pragma unroll
        for (int j = 0; j < 8; ++j) s[j] = vzero;
#pragma unroll
        for (int j = 0; j < 8; ++j) {
            int n = kblk + 16 * j + lm;
            const u16* kbase = kb + (size_t)(b * SEQ + n) * D_MODEL + h * DEPTH;
#pragma unroll
            for (int kc = 0; kc < 2; ++kc) {
                FragAB bf;
                const u16* p2 = kbase + kc * 32;
                bf.u[0] = *(const uint4*)(p2 + half * 8);
                bf.u[1] = *(const uint4*)(p2 + 16 + half * 8);
                s[j] = wmma_bf16(qf[kc], bf, s[j]);
            }
        }
        // ---- online softmax (rows r + 8*half, cols across 16 lanes of half)
        float tmax[8];
#pragma unroll
        for (int r = 0; r < 8; ++r) {
            float t = s[0][r];
#pragma unroll
            for (int j = 1; j < 8; ++j) t = fmaxf(t, s[j][r]);
            tmax[r] = t;
        }
#pragma unroll
        for (int m = 1; m < 16; m <<= 1)
#pragma unroll
            for (int r = 0; r < 8; ++r) tmax[r] = fmaxf(tmax[r], __shfl_xor(tmax[r], m, 32));
#pragma unroll
        for (int r = 0; r < 8; ++r) {
            float mn = fmaxf(mrow[r], tmax[r]);
            alpha[r] = __expf(mrow[r] - mn);
            mrow[r] = mn;
        }
        float rsum[8];
#pragma unroll
        for (int r = 0; r < 8; ++r) rsum[r] = 0.f;
#pragma unroll
        for (int j = 0; j < 8; ++j)
#pragma unroll
            for (int r = 0; r < 8; ++r) {
                float p = __expf(s[j][r] - mrow[r]);
                rsum[r] += p;
                pl[(r + 8 * half) * 136 + 16 * j + lm] = f2bf(p);
            }
#pragma unroll
        for (int m = 1; m < 16; m <<= 1)
#pragma unroll
            for (int r = 0; r < 8; ++r) rsum[r] += __shfl_xor(rsum[r], m, 32);
#pragma unroll
        for (int r = 0; r < 8; ++r) lrow[r] = lrow[r] * alpha[r] + rsum[r];
#pragma unroll
        for (int j = 0; j < 4; ++j)
#pragma unroll
            for (int r = 0; r < 8; ++r) accc[j][r] *= alpha[r];

        __syncthreads();   // P stores -> P loads (ds waitcnt via barrier)

        // ---- ctx += P @ V_blk : 4 kpos chunks x 4 depth tiles
#pragma unroll
        for (int kc = 0; kc < 4; ++kc) {
            FragAB pf;
            const u16* pb = pl + lm * 136 + kc * 32;
            pf.u[0] = *(const uint4*)(pb + half * 8);
            pf.u[1] = *(const uint4*)(pb + 16 + half * 8);
#pragma unroll
            for (int jd = 0; jd < 4; ++jd) {
                int d = 16 * jd + lm;
                const u16* vb = vt + ((size_t)(b * NHEAD + h) * DEPTH + d) * SEQ
                                   + kblk + kc * 32;
                FragAB bf;
                bf.u[0] = *(const uint4*)(vb + half * 8);
                bf.u[1] = *(const uint4*)(vb + 16 + half * 8);
                accc[jd] = wmma_bf16(pf, bf, accc[jd]);
            }
        }
        __syncthreads();
    }

#pragma unroll
    for (int jd = 0; jd < 4; ++jd)
#pragma unroll
        for (int r = 0; r < 8; ++r) {
            float v = accc[jd][r] / lrow[r];
            int row = b * SEQ + q0 + r + 8 * half;
            int col = h * DEPTH + 16 * jd + lm;
            ctx[(size_t)row * D_MODEL + col] = f2bf(v);
        }
}

// ---------------------------------------------------------------- residual + LN
// out = LN(x + y) * g + be ; wave per row (32 lanes x 32 cols)
__global__ __launch_bounds__(256) void ln_kernel(
    const float* __restrict__ x, const float* __restrict__ y,
    const float* __restrict__ g, const float* __restrict__ be,
    float* __restrict__ out, u16* __restrict__ outb)
{
    const int w = threadIdx.x >> 5, lane = threadIdx.x & 31;
    const int row = blockIdx.x * 8 + w;
    const float* xr = x + (size_t)row * D_MODEL;
    const float* yr = y + (size_t)row * D_MODEL;

    float v[32];
    float s = 0.f;
#pragma unroll
    for (int i = 0; i < 32; ++i) {
        float t = xr[lane + 32 * i] + yr[lane + 32 * i];
        v[i] = t; s += t;
    }
#pragma unroll
    for (int m = 1; m < 32; m <<= 1) s += __shfl_xor(s, m, 32);
    float mu = s * (1.f / D_MODEL);
    float s2 = 0.f;
#pragma unroll
    for (int i = 0; i < 32; ++i) { float d = v[i] - mu; s2 += d * d; }
#pragma unroll
    for (int m = 1; m < 32; m <<= 1) s2 += __shfl_xor(s2, m, 32);
    float inv = rsqrtf(s2 * (1.f / D_MODEL) + 1e-6f);
#pragma unroll
    for (int i = 0; i < 32; ++i) {
        int c = lane + 32 * i;
        float o = (v[i] - mu) * inv * g[c] + be[c];
        out[(size_t)row * D_MODEL + c] = o;
        if (outb) outb[(size_t)row * D_MODEL + c] = f2bf(o);
    }
}

// ---------------------------------------------------------------- launch

extern "C" void kernel_launch(void* const* d_in, const int* in_sizes, int n_in,
                              void* d_out, int out_size, void* d_ws, size_t ws_size,
                              hipStream_t stream) {
    const float* x  = (const float*)d_in[0];
    const float* wq = (const float*)d_in[1];
    const float* bq = (const float*)d_in[2];
    const float* wk = (const float*)d_in[3];
    const float* bk = (const float*)d_in[4];
    const float* wv = (const float*)d_in[5];
    const float* bv = (const float*)d_in[6];
    const float* wo = (const float*)d_in[7];
    const float* bo = (const float*)d_in[8];
    const float* w1 = (const float*)d_in[9];
    const float* b1 = (const float*)d_in[10];
    const float* w2 = (const float*)d_in[11];
    const float* b2 = (const float*)d_in[12];
    const float* g1 = (const float*)d_in[13];
    const float* be1= (const float*)d_in[14];
    const float* g2 = (const float*)d_in[15];
    const float* be2= (const float*)d_in[16];

    char* ws = (char*)d_ws;
    size_t off = 0;
    auto alloc = [&](size_t bytes) -> void* {
        void* p = ws + off;
        off = (off + bytes + 255) & ~(size_t)255;
        return p;
    };

    u16*   xb   = (u16*)  alloc((size_t)MROWS * D_MODEL * 2);
    u16*   wqT  = (u16*)  alloc((size_t)D_MODEL * D_MODEL * 2);
    u16*   wkT  = (u16*)  alloc((size_t)D_MODEL * D_MODEL * 2);
    u16*   wvT  = (u16*)  alloc((size_t)D_MODEL * D_MODEL * 2);
    u16*   woT  = (u16*)  alloc((size_t)D_MODEL * D_MODEL * 2);
    u16*   w1T  = (u16*)  alloc((size_t)DFF * D_MODEL * 2);
    u16*   w2T  = (u16*)  alloc((size_t)D_MODEL * DFF * 2);
    u16*   qbuf = (u16*)  alloc((size_t)MROWS * D_MODEL * 2);
    u16*   kbuf = (u16*)  alloc((size_t)MROWS * D_MODEL * 2);
    u16*   vbuf = (u16*)  alloc((size_t)MROWS * D_MODEL * 2);
    u16*   vT   = (u16*)  alloc((size_t)BATCH * NHEAD * DEPTH * SEQ * 2);
    u16*   ctx  = (u16*)  alloc((size_t)MROWS * D_MODEL * 2);
    float* aout = (float*)alloc((size_t)MROWS * D_MODEL * 4);
    float* o1f  = (float*)alloc((size_t)MROWS * D_MODEL * 4);
    u16*   o1b  = (u16*)  alloc((size_t)MROWS * D_MODEL * 2);
    u16*   hbuf = (u16*)  alloc((size_t)MROWS * DFF * 2);
    float* ffn  = (float*)alloc((size_t)MROWS * D_MODEL * 4);

    // -------- prep: conversions / transposes
    {
        int n = MROWS * D_MODEL;
        cvt_bf16_kernel<<<(n + 255) / 256, 256, 0, stream>>>(x, xb, n);
    }
    {
        int n = D_MODEL * D_MODEL;
        transpose_w_kernel<<<(n + 255) / 256, 256, 0, stream>>>(wq, wqT, D_MODEL, D_MODEL);
        transpose_w_kernel<<<(n + 255) / 256, 256, 0, stream>>>(wk, wkT, D_MODEL, D_MODEL);
        transpose_w_kernel<<<(n + 255) / 256, 256, 0, stream>>>(wv, wvT, D_MODEL, D_MODEL);
        transpose_w_kernel<<<(n + 255) / 256, 256, 0, stream>>>(wo, woT, D_MODEL, D_MODEL);
    }
    {
        int n = D_MODEL * DFF;
        transpose_w_kernel<<<(n + 255) / 256, 256, 0, stream>>>(w1, w1T, D_MODEL, DFF);
        transpose_w_kernel<<<(n + 255) / 256, 256, 0, stream>>>(w2, w2T, DFF, D_MODEL);
    }

    // -------- QKV projections (q pre-scaled by 1/sqrt(depth))
    dim3 blk(256);
    dim3 gproj(D_MODEL / BN, MROWS / BM);
    gemm_bf16_kernel<<<gproj, blk, 0, stream>>>(xb, wqT, bq, qbuf, MROWS, D_MODEL, D_MODEL, 0.125f, 0, 1);
    gemm_bf16_kernel<<<gproj, blk, 0, stream>>>(xb, wkT, bk, kbuf, MROWS, D_MODEL, D_MODEL, 1.0f, 0, 1);
    gemm_bf16_kernel<<<gproj, blk, 0, stream>>>(xb, wvT, bv, vbuf, MROWS, D_MODEL, D_MODEL, 1.0f, 0, 1);

    {
        int n = BATCH * NHEAD * DEPTH * SEQ;
        transpose_v_kernel<<<(n + 255) / 256, 256, 0, stream>>>(vbuf, vT);
    }

    // -------- attention
    attn_kernel<<<dim3(BATCH * NHEAD, SEQ / 128), blk, 0, stream>>>(qbuf, kbuf, vT, ctx);

    // -------- output projection
    gemm_bf16_kernel<<<gproj, blk, 0, stream>>>(ctx, woT, bo, aout, MROWS, D_MODEL, D_MODEL, 1.0f, 0, 0);

    // -------- LN1 (residual)
    ln_kernel<<<MROWS / 8, blk, 0, stream>>>(x, aout, g1, be1, o1f, o1b);

    // -------- FFN
    gemm_bf16_kernel<<<dim3(DFF / BN, MROWS / BM), blk, 0, stream>>>(o1b, w1T, b1, hbuf, MROWS, DFF, D_MODEL, 1.0f, 1, 1);
    gemm_bf16_kernel<<<gproj, blk, 0, stream>>>(hbuf, w2T, b2, ffn, MROWS, D_MODEL, DFF, 1.0f, 0, 0);

    // -------- LN2 -> output
    ln_kernel<<<MROWS / 8, blk, 0, stream>>>(o1f, ffn, g2, be2, (float*)d_out, (u16*)nullptr);

    (void)in_sizes; (void)n_in; (void)out_size; (void)ws_size;
}
